// HANModel_85701777424872
// MI455X (gfx1250) — compile-verified, matrix operand order checked by reading, not run on hardware
//
#include <hip/hip_runtime.h>
#include <hip/hip_bf16.h>

typedef __attribute__((ext_vector_type(16))) _Float16 v16h;
typedef __attribute__((ext_vector_type(8)))  _Float16 v8h;
typedef __attribute__((ext_vector_type(8)))  float    v8f;

// ---------------- constants ----------------
#define N_USER 50000
#define N_ITEM 50000
#define N_EDGE 400000L
#define IN_DIM 64
#define H1 4
#define HD1 32
#define F1 128
#define EMB 64

// ---------------- utility kernels ----------------
__global__ void k_f32_to_f16(const float* __restrict__ x, _Float16* __restrict__ y, long n) {
    long i = (long)blockIdx.x * blockDim.x + threadIdx.x;
    if (i < n) y[i] = (_Float16)x[i];
}

// W: [K,F] row-major f32  ->  Wt: [F,K] row-major f16
__global__ void k_wT16(const float* __restrict__ W, _Float16* __restrict__ Wt, int K, int F) {
    long i = (long)blockIdx.x * blockDim.x + threadIdx.x;
    if (i >= (long)K * F) return;
    int k = (int)(i / F), f = (int)(i % F);
    Wt[(size_t)f * K + k] = (_Float16)W[i];
}

__global__ void k_fill_u32(unsigned* __restrict__ p, unsigned v, long n) {
    long i = (long)blockIdx.x * blockDim.x + threadIdx.x;
    if (i < n) p[i] = v;
}

__global__ void k_bias_init(float* __restrict__ out, const float* __restrict__ b, int N, int F) {
    long i = (long)blockIdx.x * blockDim.x + threadIdx.x;
    if (i >= (long)N * F) return;
    out[i] = b[i % F];
}

__global__ void k_scale_relu(float* __restrict__ x, long n) {
    long i = (long)blockIdx.x * blockDim.x + threadIdx.x;
    if (i < n) x[i] = fmaxf(0.f, 0.5f * x[i]);
}

// ---------------- WMMA GEMM: C[M,N] = A[M,K] * Bt[N,K]^T, f16 in / f32 out ----------------
__device__ inline v16h load_frag16(const _Float16* __restrict__ rowptr, int kb) {
    v8h lo = *(const v8h*)(rowptr + kb);
    v8h hi = *(const v8h*)(rowptr + kb + 16);
    v16h a;
#pragma unroll
    for (int i = 0; i < 8; ++i) { a[i] = lo[i]; a[8 + i] = hi[i]; }
    return a;
}

// one wave (32 threads) per 16x64 output strip; K, N multiples of 32/64; M multiple of 16
__global__ void k_wmma_gemm(const _Float16* __restrict__ A, const _Float16* __restrict__ Bt,
                            float* __restrict__ C, int M, int N, int K) {
    const int lane = threadIdx.x & 31;
    const int hl = lane >> 4;     // 0: K-base +0 ; 1: K-base +8 (per ISA 16-bit A/B layout)
    const int l  = lane & 15;
    const int ntn = N >> 6;       // strips of 64 columns
    const int tm = blockIdx.x / ntn;
    const int ts = blockIdx.x % ntn;
    const int rowA = tm * 16 + l;

    v8f acc0 = {}, acc1 = {}, acc2 = {}, acc3 = {};
    for (int kk = 0; kk < K; kk += 32) {
        const int kb = kk + hl * 8;
        v16h a = load_frag16(A + (size_t)rowA * K, kb);
        v16h b0 = load_frag16(Bt + (size_t)(ts * 64 +  0 + l) * K, kb);
        v16h b1 = load_frag16(Bt + (size_t)(ts * 64 + 16 + l) * K, kb);
        v16h b2 = load_frag16(Bt + (size_t)(ts * 64 + 32 + l) * K, kb);
        v16h b3 = load_frag16(Bt + (size_t)(ts * 64 + 48 + l) * K, kb);
        acc0 = __builtin_amdgcn_wmma_f32_16x16x32_f16(false, a, false, b0, (short)0, acc0, false, false);
        acc1 = __builtin_amdgcn_wmma_f32_16x16x32_f16(false, a, false, b1, (short)0, acc1, false, false);
        acc2 = __builtin_amdgcn_wmma_f32_16x16x32_f16(false, a, false, b2, (short)0, acc2, false, false);
        acc3 = __builtin_amdgcn_wmma_f32_16x16x32_f16(false, a, false, b3, (short)0, acc3, false, false);
    }
    const int rb = tm * 16 + hl * 8;      // lanes 0-15 -> rows 0..7, lanes 16-31 -> rows 8..15
#pragma unroll
    for (int r = 0; r < 8; ++r) {
        C[(size_t)(rb + r) * N + ts * 64 +  0 + l] = acc0[r];
        C[(size_t)(rb + r) * N + ts * 64 + 16 + l] = acc1[r];
        C[(size_t)(rb + r) * N + ts * 64 + 32 + l] = acc2[r];
        C[(size_t)(rb + r) * N + ts * 64 + 48 + l] = acc3[r];
    }
}

// ---------------- attention pieces ----------------
// out[n,h] = sum_d f[n, h*D+d] * a[h,d]
__global__ void k_attn(const float* __restrict__ f, const float* __restrict__ a,
                       float* __restrict__ out, int N, int H, int D) {
    long i = (long)blockIdx.x * blockDim.x + threadIdx.x;
    if (i >= (long)N * H) return;
    int n = (int)(i / H), h = (int)(i % H);
    const float* fp = f + (size_t)n * H * D + (size_t)h * D;
    const float* ap = a + (size_t)h * D;
    float s = 0.f;
    for (int d = 0; d < D; ++d) s += fp[d] * ap[d];
    out[i] = s;
}

__device__ inline unsigned encf(float f) {
    unsigned b = __float_as_uint(f);
    return (b & 0x80000000u) ? ~b : (b | 0x80000000u);
}
__device__ inline float decf(unsigned u) {
    unsigned b = (u & 0x80000000u) ? (u ^ 0x80000000u) : ~u;
    return __uint_as_float(b);
}

__global__ void k_edge_max(const float* __restrict__ el, const float* __restrict__ er,
                           const int* __restrict__ src, const int* __restrict__ dst,
                           unsigned* __restrict__ emax, long E, int H) {
    long i = (long)blockIdx.x * blockDim.x + threadIdx.x;
    if (i >= E * H) return;
    long e = i / H; int h = (int)(i % H);
    float v = el[(size_t)src[e] * H + h] + er[(size_t)dst[e] * H + h];
    v = v > 0.f ? v : 0.2f * v;                  // leaky_relu 0.2
    atomicMax(emax + (size_t)dst[e] * H + h, encf(v));
}

__global__ void k_edge_exp(const float* __restrict__ el, const float* __restrict__ er,
                           const int* __restrict__ src, const int* __restrict__ dst,
                           const unsigned* __restrict__ emax,
                           float* __restrict__ ex, float* __restrict__ den, long E, int H) {
    long i = (long)blockIdx.x * blockDim.x + threadIdx.x;
    if (i >= E * H) return;
    long e = i / H; int h = (int)(i % H);
    float v = el[(size_t)src[e] * H + h] + er[(size_t)dst[e] * H + h];
    v = v > 0.f ? v : 0.2f * v;
    float m = decf(emax[(size_t)dst[e] * H + h]);
    if (!isfinite(m)) m = 0.f;                   // zero-in-degree safety (matches reference)
    float xv = expf(v - m);
    ex[i] = xv;
    atomicAdd(den + (size_t)dst[e] * H + h, xv);
}

// out[dst, f] += (ex/den) * fs[src, f]   (F = H*D)
__global__ void k_scatter(const float* __restrict__ fs, const float* __restrict__ ex,
                          const float* __restrict__ den,
                          const int* __restrict__ src, const int* __restrict__ dst,
                          float* __restrict__ out, long E, int H, int D) {
    long i = (long)blockIdx.x * blockDim.x + threadIdx.x;
    const int F = H * D;
    if (i >= E * F) return;
    long e = i / F; int f = (int)(i % F); int h = f / D;
    int s = src[e], d = dst[e];
    float alpha = ex[e * H + h] / den[(size_t)d * H + h];
    atomicAdd(out + (size_t)d * F + f, alpha * fs[(size_t)s * F + f]);
}

// ---------------- layernorm: one wave per row ----------------
__global__ void k_ln(float* __restrict__ x, const float* __restrict__ g,
                     const float* __restrict__ be, int N, int F) {
    int row = blockIdx.x * blockDim.y + threadIdx.y;
    if (row >= N) return;
    int lane = threadIdx.x;
    float* xr = x + (size_t)row * F;
    float s = 0.f, s2 = 0.f;
    for (int f = lane; f < F; f += 32) { float v = xr[f]; s += v; s2 += v * v; }
#pragma unroll
    for (int m = 16; m; m >>= 1) { s += __shfl_xor(s, m, 32); s2 += __shfl_xor(s2, m, 32); }
    float mean = s / F;
    float var  = s2 / F - mean * mean;
    float inv  = rsqrtf(var + 1e-5f);
    for (int f = lane; f < F; f += 32) xr[f] = (xr[f] - mean) * inv * g[f] + be[f];
}

// ---------------- host-side GAT orchestration ----------------
static inline long cdiv(long a, long b) { return (a + b - 1) / b; }

static void run_gat(const _Float16* xs16, const _Float16* xd16,
                    const float* Wsrc, const float* Wdst,
                    const float* al, const float* ar, const float* bias,
                    const int* src, const int* dst,
                    int Ns, int Nd, long E, int H, int D, int K,
                    _Float16* wtS, _Float16* wtD,
                    float* fs, float* fd, float* el, float* er,
                    unsigned* emax, float* den, float* ex,
                    float* outp, hipStream_t stream) {
    const int F = H * D;
    k_wT16<<<cdiv((long)K * F, 256), 256, 0, stream>>>(Wsrc, wtS, K, F);
    k_wT16<<<cdiv((long)K * F, 256), 256, 0, stream>>>(Wdst, wtD, K, F);
    k_wmma_gemm<<<(Ns / 16) * (F / 64), 32, 0, stream>>>(xs16, wtS, fs, Ns, F, K);
    k_wmma_gemm<<<(Nd / 16) * (F / 64), 32, 0, stream>>>(xd16, wtD, fd, Nd, F, K);
    k_attn<<<cdiv((long)Ns * H, 256), 256, 0, stream>>>(fs, al, el, Ns, H, D);
    k_attn<<<cdiv((long)Nd * H, 256), 256, 0, stream>>>(fd, ar, er, Nd, H, D);
    k_fill_u32<<<cdiv((long)Nd * H, 256), 256, 0, stream>>>(emax, 0x007FFFFFu /*enc(-inf)*/, (long)Nd * H);
    k_fill_u32<<<cdiv((long)Nd * H, 256), 256, 0, stream>>>((unsigned*)den, 0u, (long)Nd * H);
    k_bias_init<<<cdiv((long)Nd * F, 256), 256, 0, stream>>>(outp, bias, Nd, F);
    k_edge_max<<<cdiv(E * H, 256), 256, 0, stream>>>(el, er, src, dst, emax, E, H);
    k_edge_exp<<<cdiv(E * H, 256), 256, 0, stream>>>(el, er, src, dst, emax, ex, den, E, H);
    k_scatter<<<cdiv(E * H * D, 256), 256, 0, stream>>>(fs, ex, den, src, dst, outp, E, H, D);
}

extern "C" void kernel_launch(void* const* d_in, const int* in_sizes, int n_in,
                              void* d_out, int out_size, void* d_ws, size_t ws_size,
                              hipStream_t stream) {
    const float* feat_user = (const float*)d_in[0];
    const float* feat_item = (const float*)d_in[1];
    const int* rates_src = (const int*)d_in[2];
    const int* rates_dst = (const int*)d_in[3];
    const int* rb_src    = (const int*)d_in[4];
    const int* rb_dst    = (const int*)d_in[5];
    const float* r1_Wsrc = (const float*)d_in[6];
    const float* r1_Wdst = (const float*)d_in[7];
    const float* r1_al   = (const float*)d_in[8];
    const float* r1_ar   = (const float*)d_in[9];
    const float* r1_b    = (const float*)d_in[10];
    const float* b1_Wsrc = (const float*)d_in[11];
    const float* b1_Wdst = (const float*)d_in[12];
    const float* b1_al   = (const float*)d_in[13];
    const float* b1_ar   = (const float*)d_in[14];
    const float* b1_b    = (const float*)d_in[15];
    const float* r2_Wsrc = (const float*)d_in[16];
    const float* r2_Wdst = (const float*)d_in[17];
    const float* r2_al   = (const float*)d_in[18];
    const float* r2_ar   = (const float*)d_in[19];
    const float* r2_b    = (const float*)d_in[20];
    const float* b2_Wsrc = (const float*)d_in[21];
    const float* b2_Wdst = (const float*)d_in[22];
    const float* b2_al   = (const float*)d_in[23];
    const float* b2_ar   = (const float*)d_in[24];
    const float* b2_b    = (const float*)d_in[25];
    // d_in[26..29]: Wa1/ba1/Wa2/ba2 are dead: softmax of 2 scalars -> mean is exactly 0.5
    const float* g1  = (const float*)d_in[30];
    const float* be1 = (const float*)d_in[31];
    const float* g2  = (const float*)d_in[32];
    const float* be2 = (const float*)d_in[33];

    float* out = (float*)d_out;
    float* o_user = out;                               // [N_USER, EMB]
    float* o_item = out + (size_t)N_USER * EMB;        // [N_ITEM, EMB]

    // bump allocator on workspace
    char* ws = (char*)d_ws;
    size_t off = 0;
    auto alloc = [&](size_t bytes) -> void* {
        void* p = ws + off;
        off += (bytes + 255) & ~(size_t)255;
        return p;
    };

    // persistent across phases
    float* h_user = (float*)alloc((size_t)N_USER * F1 * 4);
    float* h_item = (float*)alloc((size_t)N_ITEM * F1 * 4);
    const size_t phase_base = off;

    // ---------------- phase 1 ----------------
    _Float16* xu16 = (_Float16*)alloc((size_t)N_USER * IN_DIM * 2);
    _Float16* xi16 = (_Float16*)alloc((size_t)N_ITEM * IN_DIM * 2);
    _Float16* wtSr1 = (_Float16*)alloc((size_t)IN_DIM * F1 * 2);
    _Float16* wtDr1 = (_Float16*)alloc((size_t)IN_DIM * F1 * 2);
    _Float16* wtSb1 = (_Float16*)alloc((size_t)IN_DIM * F1 * 2);
    _Float16* wtDb1 = (_Float16*)alloc((size_t)IN_DIM * F1 * 2);
    float* fs_r = (float*)alloc((size_t)N_USER * F1 * 4);
    float* fd_r = (float*)alloc((size_t)N_ITEM * F1 * 4);
    float* fs_b = (float*)alloc((size_t)N_ITEM * F1 * 4);
    float* fd_b = (float*)alloc((size_t)N_USER * F1 * 4);
    float* el_r = (float*)alloc((size_t)N_USER * H1 * 4);
    float* er_r = (float*)alloc((size_t)N_ITEM * H1 * 4);
    float* el_b = (float*)alloc((size_t)N_ITEM * H1 * 4);
    float* er_b = (float*)alloc((size_t)N_USER * H1 * 4);
    unsigned* emax_r = (unsigned*)alloc((size_t)N_ITEM * H1 * 4);
    float*    den_r  = (float*)alloc((size_t)N_ITEM * H1 * 4);
    unsigned* emax_b = (unsigned*)alloc((size_t)N_USER * H1 * 4);
    float*    den_b  = (float*)alloc((size_t)N_USER * H1 * 4);
    float* ex_r = (float*)alloc((size_t)N_EDGE * H1 * 4);
    float* ex_b = (float*)alloc((size_t)N_EDGE * H1 * 4);

    k_f32_to_f16<<<cdiv((long)N_USER * IN_DIM, 256), 256, 0, stream>>>(feat_user, xu16, (long)N_USER * IN_DIM);
    k_f32_to_f16<<<cdiv((long)N_ITEM * IN_DIM, 256), 256, 0, stream>>>(feat_item, xi16, (long)N_ITEM * IN_DIM);

    // h_item = GAT_rates(user->item), h_user = GAT_rated_by(item->user)
    run_gat(xu16, xi16, r1_Wsrc, r1_Wdst, r1_al, r1_ar, r1_b, rates_src, rates_dst,
            N_USER, N_ITEM, N_EDGE, H1, HD1, IN_DIM,
            wtSr1, wtDr1, fs_r, fd_r, el_r, er_r, emax_r, den_r, ex_r, h_item, stream);
    run_gat(xi16, xu16, b1_Wsrc, b1_Wdst, b1_al, b1_ar, b1_b, rb_src, rb_dst,
            N_ITEM, N_USER, N_EDGE, H1, HD1, IN_DIM,
            wtSb1, wtDb1, fs_b, fd_b, el_b, er_b, emax_b, den_b, ex_b, h_user, stream);

    // sem -> *0.5 relu ; layernorm (in place)
    k_scale_relu<<<cdiv((long)N_USER * F1, 256), 256, 0, stream>>>(h_user, (long)N_USER * F1);
    k_scale_relu<<<cdiv((long)N_ITEM * F1, 256), 256, 0, stream>>>(h_item, (long)N_ITEM * F1);
    k_ln<<<cdiv(N_USER, 8), dim3(32, 8), 0, stream>>>(h_user, g1, be1, N_USER, F1);
    k_ln<<<cdiv(N_ITEM, 8), dim3(32, 8), 0, stream>>>(h_item, g1, be1, N_ITEM, F1);

    // ---------------- phase 2 (reuse phase-1 scratch) ----------------
    off = phase_base;
    _Float16* hu16 = (_Float16*)alloc((size_t)N_USER * F1 * 2);
    _Float16* hi16 = (_Float16*)alloc((size_t)N_ITEM * F1 * 2);
    _Float16* wtS2 = (_Float16*)alloc((size_t)F1 * EMB * 2);
    _Float16* wtD2 = (_Float16*)alloc((size_t)F1 * EMB * 2);
    _Float16* wtS3 = (_Float16*)alloc((size_t)F1 * EMB * 2);
    _Float16* wtD3 = (_Float16*)alloc((size_t)F1 * EMB * 2);
    float* fs2_r = (float*)alloc((size_t)N_USER * EMB * 4);
    float* fd2_r = (float*)alloc((size_t)N_ITEM * EMB * 4);
    float* fs2_b = (float*)alloc((size_t)N_ITEM * EMB * 4);
    float* fd2_b = (float*)alloc((size_t)N_USER * EMB * 4);
    float* el2_r = (float*)alloc((size_t)N_USER * 4);
    float* er2_r = (float*)alloc((size_t)N_ITEM * 4);
    float* el2_b = (float*)alloc((size_t)N_ITEM * 4);
    float* er2_b = (float*)alloc((size_t)N_USER * 4);
    unsigned* emax2_r = (unsigned*)alloc((size_t)N_ITEM * 4);
    float*    den2_r  = (float*)alloc((size_t)N_ITEM * 4);
    unsigned* emax2_b = (unsigned*)alloc((size_t)N_USER * 4);
    float*    den2_b  = (float*)alloc((size_t)N_USER * 4);
    float* ex2_r = (float*)alloc((size_t)N_EDGE * 4);
    float* ex2_b = (float*)alloc((size_t)N_EDGE * 4);

    k_f32_to_f16<<<cdiv((long)N_USER * F1, 256), 256, 0, stream>>>(h_user, hu16, (long)N_USER * F1);
    k_f32_to_f16<<<cdiv((long)N_ITEM * F1, 256), 256, 0, stream>>>(h_item, hi16, (long)N_ITEM * F1);

    // o_item = GAT_rates(user->item), o_user = GAT_rated_by(item->user); outputs go straight to d_out
    run_gat(hu16, hi16, r2_Wsrc, r2_Wdst, r2_al, r2_ar, r2_b, rates_src, rates_dst,
            N_USER, N_ITEM, N_EDGE, 1, EMB, F1,
            wtS2, wtD2, fs2_r, fd2_r, el2_r, er2_r, emax2_r, den2_r, ex2_r, o_item, stream);
    run_gat(hi16, hu16, b2_Wsrc, b2_Wdst, b2_al, b2_ar, b2_b, rb_src, rb_dst,
            N_ITEM, N_USER, N_EDGE, 1, EMB, F1,
            wtS3, wtD3, fs2_b, fd2_b, el2_b, er2_b, emax2_b, den2_b, ex2_b, o_user, stream);

    // sem -> *0.5 relu ; layernorm (in place on d_out)
    k_scale_relu<<<cdiv((long)N_USER * EMB, 256), 256, 0, stream>>>(o_user, (long)N_USER * EMB);
    k_scale_relu<<<cdiv((long)N_ITEM * EMB, 256), 256, 0, stream>>>(o_item, (long)N_ITEM * EMB);
    k_ln<<<cdiv(N_USER, 8), dim3(32, 8), 0, stream>>>(o_user, g2, be2, N_USER, EMB);
    k_ln<<<cdiv(N_ITEM, 8), dim3(32, 8), 0, stream>>>(o_item, g2, be2, N_ITEM, EMB);
}
